// DynamicGraphIpaFrameDirichletDenoiser_81432579932303
// MI455X (gfx1250) — matrix operand order repaired and checked by reading, not compiled, not run on hardware
//
#include <hip/hip_runtime.h>
#include <hip/hip_bf16.h>
#include <math.h>

// ---------------------------------------------------------------------------
// CDNA5 / gfx1250, wave32.  GEMM-dominated graph IPA denoiser.
// All matmuls -> v_wmma_f32_16x16x32_bf16; A-tiles streamed to LDS with
// global_load_async_to_lds_b128 (ASYNCcnt), bf16 operands pre-converted.
// ---------------------------------------------------------------------------

#ifndef DISABLE_ASYNC_LDS
#define USE_ASYNC_LDS 1
#endif

typedef __attribute__((ext_vector_type(16))) __bf16 v16bf;
typedef __attribute__((ext_vector_type(8)))  __bf16 v8bf;
typedef __attribute__((ext_vector_type(8)))  float  v8f;

#define NNODE   3072
#define CS      256
#define CZ      128
#define CH      256
#define NH      8
#define PQn     8
#define PVn     12
#define KNN_K   20
#define LR_K    40
#define EPN_SP  60
#define EPN_SEQ 4
#define N_AA    20
#define NODE_IN 624
#define EIN     712
#define EF_RAW_D 200          // 64 rbf + 64 sinusoid + 4 quat + 68 pad
#define OUT_FEAT 3456         // NH*(CZ + CH + PVn*4)

#define TILE_M 128
#define TILE_N 128
#define TILE_K 32
#define LDSK   40             // padded LDS stride (bf16 elems), keeps 16B align

// ========================= bf16 convert kernels ============================
__global__ void k_cvt_bf16_pad(const float* __restrict__ x, __bf16* __restrict__ y,
                               long rows, int K, int Kp) {
  long idx = blockIdx.x * (long)blockDim.x + threadIdx.x;
  if (idx >= rows * (long)Kp) return;
  long r = idx / Kp; int d = (int)(idx % Kp);
  y[idx] = (d < K) ? (__bf16)x[r * (long)K + d] : (__bf16)0.f;
}

__global__ void k_cvt_bf16(const float* __restrict__ x, __bf16* __restrict__ y, long n) {
  long idx = blockIdx.x * (long)blockDim.x + threadIdx.x;
  if (idx < n) y[idx] = (__bf16)x[idx];
}

// ========================= WMMA GEMM =======================================
// C[M,N] = act( A[M,Kp(bf16)] @ W[K,N(bf16)] + bias[N] )
__global__ __launch_bounds__(256)
void k_gemm_wmma(const __bf16* __restrict__ A, const __bf16* __restrict__ W,
                 const float* __restrict__ bias, float* __restrict__ C,
                 int M, int Nn, int K, int Kp, int act)
{
  __shared__ __align__(16) __bf16 la[TILE_M * LDSK];
  __shared__ __align__(16) __bf16 lb[TILE_N * LDSK];   // stored [n][k]

  const int tid    = threadIdx.x;
  const int lane   = tid & 31;
  const int wid    = tid >> 5;
  const int laneLo = lane & 15;
  const int laneHi = lane >> 4;
  const int m0 = blockIdx.y * TILE_M;
  const int n0 = blockIdx.x * TILE_N;
  const int wm = (wid >> 2) * 64;   // wave M offset within tile
  const int wn = (wid & 3) * 32;    // wave N offset within tile

  v8f acc[4][2];
#pragma unroll
  for (int a = 0; a < 4; ++a)
#pragma unroll
    for (int b = 0; b < 2; ++b)
#pragma unroll
      for (int r = 0; r < 8; ++r) acc[a][b][r] = 0.f;

  const int bk = tid >> 3;           // 0..31  (B: k within tile)
  const int bn = (tid & 7) * 16;     // 0..112 (B: n chunk base)

  for (int k0 = 0; k0 < K; k0 += TILE_K) {
    // ---- stage A tile: 512 chunks of 8 bf16 (16B), 2 per thread ----
#pragma unroll
    for (int cc = 0; cc < 2; ++cc) {
      int c   = tid + cc * 256;
      int row = c >> 2;              // 0..127
      int col = (c & 3) * 8;         // 0,8,16,24
      int gm  = m0 + row;
      __bf16* dst = &la[row * LDSK + col];
      if (gm < M && (k0 + col + 8) <= Kp) {
        const __bf16* src = A + (size_t)gm * Kp + k0 + col;
#ifdef USE_ASYNC_LDS
        // per-lane 16B global->LDS DMA, tracked with ASYNCcnt
        unsigned dlds = (unsigned)(size_t)dst;   // low 32 bits = LDS offset
        asm volatile("global_load_async_to_lds_b128 %0, %1, off"
                     :: "v"(dlds), "v"(src) : "memory");
#else
        *(v8bf*)dst = *(const v8bf*)src;
#endif
      } else {
        int rowc = (gm < M) ? gm : (M - 1);
        const __bf16* base = A + (size_t)rowc * Kp;
#pragma unroll
        for (int t = 0; t < 8; ++t) {        // clamp+select, branch-free
          int kk  = k0 + col + t;
          int kkc = (kk < Kp) ? kk : (Kp - 1);
          __bf16 v = base[kkc];
          if (gm >= M || kk >= Kp) v = (__bf16)0.f;
          dst[t] = v;
        }
      }
    }
    // ---- stage B tile transposed into LDS, clamp+select ----
    {
      int gk  = k0 + bk;
      int gkc = (gk < K) ? gk : (K - 1);
      const __bf16* Wp = W + (size_t)gkc * Nn;
#pragma unroll
      for (int t = 0; t < 16; ++t) {
        int nn  = n0 + bn + t;
        int nnc = (nn < Nn) ? nn : (Nn - 1);
        __bf16 v = Wp[nnc];
        if (gk >= K || nn >= Nn) v = (__bf16)0.f;
        lb[(bn + t) * LDSK + bk] = v;
      }
      if (gk + TILE_K < K)
        __builtin_prefetch(Wp + (size_t)TILE_K * Nn + n0 + bn, 0, 1);
    }
#ifdef USE_ASYNC_LDS
    asm volatile("s_wait_asynccnt 0" ::: "memory");
#endif
    __syncthreads();

    // ---- fragments (ISA 16-bit A/B VGPR layouts) ----
    v16bf afr[4], bfr[2];
#pragma unroll
    for (int t = 0; t < 4; ++t) {
      const __bf16* p = &la[(wm + t*16 + laneLo) * LDSK + laneHi * 8];
      union { v16bf v; v8bf h[2]; } u;
      u.h[0] = *(const v8bf*)p;
      u.h[1] = *(const v8bf*)(p + 16);
      afr[t] = u.v;
    }
#pragma unroll
    for (int t = 0; t < 2; ++t) {
      const __bf16* p = &lb[(wn + t*16 + laneLo) * LDSK + laneHi * 8];
      union { v16bf v; v8bf h[2]; } u;
      u.h[0] = *(const v8bf*)p;
      u.h[1] = *(const v8bf*)(p + 16);
      bfr[t] = u.v;
    }
#pragma unroll
    for (int mt = 0; mt < 4; ++mt)
#pragma unroll
      for (int nt = 0; nt < 2; ++nt)
        acc[mt][nt] = __builtin_amdgcn_wmma_f32_16x16x32_bf16(
            false, afr[mt], false, bfr[nt], (short)0, acc[mt][nt], false, false);
    __syncthreads();
  }

  // ---- epilogue: f32 C/D layout (VGPR r -> M=r / M=8+r per lane half) ----
#pragma unroll
  for (int mt = 0; mt < 4; ++mt)
#pragma unroll
    for (int nt = 0; nt < 2; ++nt)
#pragma unroll
      for (int r = 0; r < 8; ++r) {
        int row = m0 + wm + mt*16 + laneHi*8 + r;
        int col = n0 + wn + nt*16 + laneLo;
        if (row < M && col < Nn) {
          float v = acc[mt][nt][r];
          if (bias) v += bias[col];
          if (act)  v = fmaxf(v, 0.f);
          C[(size_t)row * Nn + col] = v;
        }
      }
}

// ========================= helpers =========================================
__device__ inline void quat_rot(const float* q, float R[3][3]) {
  float w = q[0], x = q[1], y = q[2], z = q[3];
  R[0][0] = 1.f - 2.f*(y*y + z*z); R[0][1] = 2.f*(x*y - w*z); R[0][2] = 2.f*(x*z + w*y);
  R[1][0] = 2.f*(x*y + w*z); R[1][1] = 1.f - 2.f*(x*x + z*z); R[1][2] = 2.f*(y*z - w*x);
  R[2][0] = 2.f*(x*z - w*y); R[2][1] = 2.f*(y*z + w*x); R[2][2] = 1.f - 2.f*(x*x + y*y);
}

__global__ void k_zero(float* p, long n) {
  long i = blockIdx.x * (long)blockDim.x + threadIdx.x;
  if (i < n) p[i] = 0.f;
}

__global__ void k_quat_norm(const float* rig, float* quats, float* trans) {
  int n = blockIdx.x * blockDim.x + threadIdx.x;
  if (n >= NNODE) return;
  float q0 = rig[n*7+0], q1 = rig[n*7+1], q2 = rig[n*7+2], q3 = rig[n*7+3];
  float inv = rsqrtf(q0*q0 + q1*q1 + q2*q2 + q3*q3);
  quats[n*4+0] = q0*inv; quats[n*4+1] = q1*inv;
  quats[n*4+2] = q2*inv; quats[n*4+3] = q3*inv;
  trans[n*3+0] = rig[n*7+4]; trans[n*3+1] = rig[n*7+5]; trans[n*3+2] = rig[n*7+6];
}

__global__ void k_center_reduce(const float* trans, const float* mask, float* acc) {
  int n = blockIdx.x * blockDim.x + threadIdx.x;
  if (n >= NNODE) return;
  float m = mask[n];
  atomicAdd(&acc[0], trans[n*3+0]*m);
  atomicAdd(&acc[1], trans[n*3+1]*m);
  atomicAdd(&acc[2], trans[n*3+2]*m);
  atomicAdd(&acc[3], m);
}

__global__ void k_center_apply(float* trans, const float* acc) {
  int i = blockIdx.x * blockDim.x + threadIdx.x;
  if (i >= NNODE*3) return;
  int c = i % 3;
  float center = acc[c] / (acc[3] + 1e-8f);
  trans[i] = (trans[i] - center) * 0.1f;   // layers operate on 0.1-scaled trans
}

__global__ void k_node_input(const float* seqp, const float* nmask,
                             const float* time_w, const float* t, float* out) {
  long idx = blockIdx.x * (long)blockDim.x + threadIdx.x;
  if (idx >= (long)NNODE * NODE_IN) return;
  int n = (int)(idx / NODE_IN), d = (int)(idx % NODE_IN);
  float v = 0.f;
  const float nl = -9.2103403719761836f;  // -ln(10000)
  if (d < 128) {
    float fr = expf((2.f*d) * (nl / 256.f)); v = cosf((float)n * fr);
  } else if (d < 256) {
    float fr = expf((2.f*(d-128)) * (nl / 256.f)); v = sinf((float)n * fr);
  } else if (d < 288) {
    v = cosf(6.2831853071795865f * time_w[d-256] * t[0]);
  } else if (d < 320) {
    v = sinf(6.2831853071795865f * time_w[d-288] * t[0]);
  } else if (d < 340) {
    v = seqp[(size_t)n*N_AA + (d-320)];
  } else if (d == 340) {
    v = nmask[n];
  }
  out[idx] = v;
}

__global__ void k_seq_build(int* dst) {
  int idx = blockIdx.x * blockDim.x + threadIdx.x;
  if (idx >= NNODE * EPN_SEQ) return;
  int n = idx / EPN_SEQ, s = idx % EPN_SEQ;
  int off = (s == 0) ? -2 : (s == 1) ? -1 : (s == 2) ? 1 : 2;
  int j = n + off;
  dst[idx] = (j >= 0 && j < NNODE) ? j : -1;
}

__global__ void k_knn(const float* trans, int* edst, int layer) {
  int i = blockIdx.x * blockDim.x + threadIdx.x;
  if (i >= NNODE) return;
  float tx = trans[i*3], ty = trans[i*3+1], tz = trans[i*3+2];
  float bd[KNN_K]; int bj[KNN_K];
  for (int k = 0; k < KNN_K; ++k) { bd[k] = 3.0e38f; bj[k] = i; }
  for (int j = 0; j < NNODE; ++j) {
    if (j == i) continue;
    float dx = trans[j*3]-tx, dy = trans[j*3+1]-ty, dz = trans[j*3+2]-tz;
    float d2 = dx*dx + dy*dy + dz*dz;
    if (d2 < bd[KNN_K-1]) {
      int p = KNN_K - 1;
      while (p > 0 && bd[p-1] > d2) { bd[p] = bd[p-1]; bj[p] = bj[p-1]; --p; }
      bd[p] = d2; bj[p] = j;
    }
  }
  for (int k = 0; k < KNN_K; ++k) edst[i*EPN_SP + k] = bj[k];
  // long-range edges: hash-sampled (Gumbel top-k RNG not reproducible here)
  unsigned base = (unsigned)(i * 2654435761u) ^ (unsigned)(layer * 0x9E3779B9u);
  for (int k = 0; k < LR_K; ++k) {
    unsigned h = base ^ (unsigned)(k * 0x85EBCA6Bu);
    h ^= h >> 16; h *= 0x7FEB352Du; h ^= h >> 15; h *= 0x846CA68Bu; h ^= h >> 16;
    int j = (int)(h % (unsigned)NNODE);
    if (j == i) j = (j + 1) % NNODE;
    edst[i*EPN_SP + KNN_K + k] = j;
  }
}

// rbf(64) | sinusoid(j-i,64) | quat_rel(4) | zeros(68)
__global__ void k_edge_feats(const float* quats, const float* trans,
                             const int* edst, int EPN, long E, float* out) {
  long idx = blockIdx.x * (long)blockDim.x + threadIdx.x;
  if (idx >= E * EF_RAW_D) return;
  long e = idx / EF_RAW_D; int d = (int)(idx % EF_RAW_D);
  int i = (int)(e / EPN); int j = edst[e];
  float v = 0.f;
  if (j >= 0) {
    if (d < 64) {
      float dx = (trans[i*3]-trans[j*3])*10.f;
      float dy = (trans[i*3+1]-trans[j*3+1])*10.f;
      float dz = (trans[i*3+2]-trans[j*3+2])*10.f;
      float dist = sqrtf(dx*dx + dy*dy + dz*dz + 1e-10f);
      float mu = 20.f * d / 63.f;
      float u = (dist - mu) / 0.3125f;
      v = expf(-u*u);
    } else if (d < 128) {
      int k = (d - 64) & 31; bool isSin = d >= 96;
      float fr = expf((2.f*k) * (-9.2103403719761836f / 64.f));
      float a = (float)(j - i) * fr;
      v = isSin ? sinf(a) : cosf(a);
    } else if (d < 132) {
      const float* qi = quats + i*4; const float* qj = quats + j*4;
      float aw = qi[0], ax = -qi[1], ay = -qi[2], az = -qi[3];
      float bw = qj[0], bx = qj[1], by = qj[2], bz = qj[3];
      float r[4];
      r[0] = aw*bw - ax*bx - ay*by - az*bz;
      r[1] = aw*bx + ax*bw + ay*bz - az*by;
      r[2] = aw*by - ax*bz + ay*bw + az*bx;
      r[3] = aw*bz + ax*by - ay*bx + az*bw;
      v = r[d - 128];
    }
  }
  out[idx] = v;
}

// out[e] = [ base[e](D0) | X[i](D1) | X[j](D1) ]
__global__ void k_gather_concat(const float* base, int D0, const float* X, int D1,
                                const int* edst, int EPN, long E, float* out) {
  long idx = blockIdx.x * (long)blockDim.x + threadIdx.x;
  int Dt = D0 + 2*D1;
  if (idx >= E * Dt) return;
  long e = idx / Dt; int d = (int)(idx % Dt);
  float v;
  if (d < D0) {
    v = base[e*D0 + d];
  } else if (d < D0 + D1) {
    int i = (int)(e / EPN);
    v = X[(size_t)i*D1 + (d - D0)];
  } else {
    int j = edst[e];
    v = (j >= 0) ? X[(size_t)j*D1 + (d - D0 - D1)] : 0.f;
  }
  out[idx] = v;
}

// out = postscale * LN( residual + premul * x ) * g + b  (nullable extras)
__global__ __launch_bounds__(256)
void k_layernorm(const float* x, const float* res, const float* premul,
                 const float* g, const float* b, const float* postscale,
                 float* out, int D) {
  int row = blockIdx.x, tid = threadIdx.x;
  __shared__ float red[256];
  float pm = premul ? premul[row] : 1.f;
  int d0 = tid, d1 = tid + 256;
  float v0 = 0.f, v1 = 0.f;
  if (d0 < D) { v0 = x[(size_t)row*D + d0] * pm; if (res) v0 += res[(size_t)row*D + d0]; }
  if (d1 < D) { v1 = x[(size_t)row*D + d1] * pm; if (res) v1 += res[(size_t)row*D + d1]; }
  red[tid] = v0 + v1; __syncthreads();
  for (int o = 128; o > 0; o >>= 1) { if (tid < o) red[tid] += red[tid + o]; __syncthreads(); }
  float mean = red[0] / D; __syncthreads();
  float c0 = (d0 < D) ? v0 - mean : 0.f;
  float c1 = (d1 < D) ? v1 - mean : 0.f;
  red[tid] = c0*c0 + c1*c1; __syncthreads();
  for (int o = 128; o > 0; o >>= 1) { if (tid < o) red[tid] += red[tid + o]; __syncthreads(); }
  float rstd = rsqrtf(red[0] / D + 1e-5f);
  float ps = postscale ? postscale[row] : 1.f;
  if (d0 < D) out[(size_t)row*D + d0] = (c0 * rstd * g[d0] + b[d0]) * ps;
  if (d1 < D) out[(size_t)row*D + d1] = (c1 * rstd * g[d1] + b[d1]) * ps;
}

__global__ void k_pts_glob(const float* quats, const float* trans,
                           const float* loc, float* glob, int HP) {
  int idx = blockIdx.x * blockDim.x + threadIdx.x;
  if (idx >= NNODE * HP) return;
  int n = idx / HP, p = idx % HP;
  float R[3][3]; quat_rot(quats + n*4, R);
  const float* l = loc + (size_t)n*HP*3 + p*3;
  float x = l[0], y = l[1], z = l[2];
  float* o = glob + (size_t)n*HP*3 + p*3;
  o[0] = R[0][0]*x + R[0][1]*y + R[0][2]*z + trans[n*3+0];
  o[1] = R[1][0]*x + R[1][1]*y + R[1][2]*z + trans[n*3+1];
  o[2] = R[2][0]*x + R[2][1]*y + R[2][2]*z + trans[n*3+2];
}

__global__ void k_ipa_logits(const float* q, const float* k, const float* qpg,
                             const float* kpg, const float* bze, const float* gamma,
                             const float* mask, const int* edst, int EPN, long E,
                             float* logits) {
  long idx = blockIdx.x * (long)blockDim.x + threadIdx.x;
  if (idx >= E * NH) return;
  long e = idx / NH; int h = (int)(idx % NH);
  int i = (int)(e / EPN); int j = edst[e];
  if (j < 0) { logits[idx] = -1e30f; return; }
  const float* qi = q + (size_t)i*(NH*CH) + h*CH;
  const float* kj = k + (size_t)j*(NH*CH) + h*CH;
  float qk = 0.f;
  for (int c = 0; c < CH; ++c) qk += qi[c] * kj[c];
  const float* qp = qpg + (size_t)i*(NH*PQn*3) + h*PQn*3;
  const float* kp = kpg + (size_t)j*(NH*PQn*3) + h*PQn*3;
  float ptsq = 0.f;
  for (int p = 0; p < PQn*3; ++p) { float d = qp[p] - kp[p]; ptsq += d*d; }
  float g = gamma[h];
  float sp = (g > 20.f) ? g : log1pf(expf(g));
  float l = (qk * 0.0625f + bze[e*NH + h] - 0.5f * 0.16666667f * sp * ptsq) * 0.57735027f;
  l += (mask[j] - 1.f) * 1e9f;
  logits[idx] = l;
}

__global__ void k_ipa_softmax(float* logits, int EPN) {
  int idx = blockIdx.x * blockDim.x + threadIdx.x;
  if (idx >= NNODE * NH) return;
  int n = idx / NH, h = idx % NH;
  float mx = -3e38f;
  for (int e = 0; e < EPN; ++e)
    mx = fmaxf(mx, logits[((size_t)n*EPN + e)*NH + h]);
  float den = 0.f;
  for (int e = 0; e < EPN; ++e)
    den += expf(logits[((size_t)n*EPN + e)*NH + h] - mx);
  float inv = 1.f / (den + 1e-9f);
  for (int e = 0; e < EPN; ++e) {
    size_t o = ((size_t)n*EPN + e)*NH + h;
    logits[o] = expf(logits[o] - mx) * inv;
  }
}

__global__ __launch_bounds__(256)
void k_ipa_accum(const float* a, const float* v, const float* vpg, const float* z,
                 const int* edst, const float* quats, const float* trans,
                 int EPN, float* feat) {
  int n = blockIdx.x, tid = threadIdx.x;
  __shared__ float as[NH * EPN_SP];
  __shared__ int js[EPN_SP];
  for (int e = tid; e < EPN; e += 256) js[e] = edst[(size_t)n*EPN + e];
  for (int t = tid; t < NH*EPN; t += 256) {
    int e = t / NH, h = t % NH;
    as[h*EPN_SP + e] = a[((size_t)n*EPN + e)*NH + h];
  }
  __syncthreads();
  float* f = feat + (size_t)n * OUT_FEAT;
  // o = sum a * v[j]
  for (int t = tid; t < NH*CH; t += 256) {
    int h = t / CH, c = t % CH;
    float s = 0.f;
    for (int e = 0; e < EPN; ++e) {
      int j = js[e]; if (j < 0) continue;
      s += as[h*EPN_SP + e] * v[(size_t)j*(NH*CH) + h*CH + c];
    }
    f[t] = s;
  }
  // opt: point sums -> local frame + norm
  if (tid < NH*PVn) {
    int h = tid / PVn, p = tid % PVn;
    float gx = 0.f, gy = 0.f, gz = 0.f;
    for (int e = 0; e < EPN; ++e) {
      int j = js[e]; if (j < 0) continue;
      float w = as[h*EPN_SP + e];
      const float* vp = vpg + (size_t)j*(NH*PVn*3) + (h*PVn + p)*3;
      gx += w*vp[0]; gy += w*vp[1]; gz += w*vp[2];
    }
    float R[3][3]; quat_rot(quats + n*4, R);
    float dx = gx - trans[n*3], dy = gy - trans[n*3+1], dz = gz - trans[n*3+2];
    float lx = R[0][0]*dx + R[1][0]*dy + R[2][0]*dz;   // R^T
    float ly = R[0][1]*dx + R[1][1]*dy + R[2][1]*dz;
    float lz = R[0][2]*dx + R[1][2]*dy + R[2][2]*dz;
    int o = NH*CH + (h*PVn + p)*3;
    f[o] = lx; f[o+1] = ly; f[o+2] = lz;
    f[NH*CH + NH*PVn*3 + h*PVn + p] = sqrtf(lx*lx + ly*ly + lz*lz + 1e-8f);
  }
  // opair = sum a * z_e
  int base = NH*CH + NH*PVn*4;
  for (int t = tid; t < NH*CZ; t += 256) {
    int h = t / CZ, c = t % CZ;
    float s = 0.f;
    for (int e = 0; e < EPN; ++e) {
      if (js[e] < 0) continue;
      s += as[h*EPN_SP + e] * z[((size_t)n*EPN + e)*CZ + c];
    }
    f[base + t] = s;
  }
}

__global__ void k_rowmul(const float* x, const float* scale, float* out, int D) {
  long idx = blockIdx.x * (long)blockDim.x + threadIdx.x;
  if (idx >= (long)NNODE * D) return;
  out[idx] = x[idx] * scale[idx / D];
}

__global__ void k_update_rigids(const float* bb, const float* nmask,
                                float* quats, float* trans) {
  int n = blockIdx.x * blockDim.x + threadIdx.x;
  if (n >= NNODE) return;
  float m = nmask[n];
  float b0 = bb[n*6+0]*m, b1 = bb[n*6+1]*m, b2 = bb[n*6+2]*m;
  float b3 = bb[n*6+3]*m, b4 = bb[n*6+4]*m, b5 = bb[n*6+5]*m;
  float dw = 1.f, dx = b0, dy = b1, dz = b2;
  float inv = rsqrtf(dw*dw + dx*dx + dy*dy + dz*dz);
  dw *= inv; dx *= inv; dy *= inv; dz *= inv;
  float aw = quats[n*4+0], ax = quats[n*4+1], ay = quats[n*4+2], az = quats[n*4+3];
  float R[3][3]; quat_rot(quats + n*4, R);   // rotation from OLD quats
  quats[n*4+0] = aw*dw - ax*dx - ay*dy - az*dz;
  quats[n*4+1] = aw*dx + ax*dw + ay*dz - az*dy;
  quats[n*4+2] = aw*dy - ax*dz + ay*dw + az*dx;
  quats[n*4+3] = aw*dz + ax*dy - ay*dx + az*dw;
  trans[n*3+0] += R[0][0]*b3 + R[0][1]*b4 + R[0][2]*b5;
  trans[n*3+1] += R[1][0]*b3 + R[1][1]*b4 + R[1][2]*b5;
  trans[n*3+2] += R[2][0]*b3 + R[2][1]*b4 + R[2][2]*b5;
}

__global__ void k_write_rigids(const float* quats, const float* trans, float* out) {
  int n = blockIdx.x * blockDim.x + threadIdx.x;
  if (n >= NNODE) return;
  out[n*7+0] = quats[n*4+0]; out[n*7+1] = quats[n*4+1];
  out[n*7+2] = quats[n*4+2]; out[n*7+3] = quats[n*4+3];
  out[n*7+4] = trans[n*3+0]*10.f; out[n*7+5] = trans[n*3+1]*10.f;
  out[n*7+6] = trans[n*3+2]*10.f;
}

__global__ void k_concat2(const float* a, int Da, const float* b, int Db, float* out) {
  long idx = blockIdx.x * (long)blockDim.x + threadIdx.x;
  int Dt = Da + Db;
  if (idx >= (long)NNODE * Dt) return;
  int n = (int)(idx / Dt), d = (int)(idx % Dt);
  out[idx] = (d < Da) ? a[(size_t)n*Da + d] : b[(size_t)n*Db + (d - Da)];
}

// ========================= host orchestration ==============================
struct Lin { const float* w; const float* b; };
struct Mlp3 { Lin l0, l1, l2; };
struct LnP { const float* g; const float* b; };
struct IpaP { Lin q, k, v, qp, kp, vp, bz; const float* gamma; Lin out; };
struct LayerP {
  Mlp3 ee; LnP eeln; Mlp3 seu; LnP seln; IpaP sp; LnP ln1;
  Lin etn; Lin et0, et1; LnP etln; IpaP sq; LnP ln2; Lin bb; Mlp3 nt; LnP ntln;
};

extern "C" void kernel_launch(void* const* d_in, const int* in_sizes, int n_in,
                              void* d_out, int out_size, void* d_ws, size_t ws_size,
                              hipStream_t stream) {
  (void)in_sizes; (void)n_in; (void)out_size; (void)ws_size;

  const float* rigids = (const float*)d_in[0];
  const float* res_mask = (const float*)d_in[1];
  const float* noise_mask = (const float*)d_in[2];
  const float* seq_probs = (const float*)d_in[3];
  const float* tptr = (const float*)d_in[4];

  int pi = 5;
  auto P = [&]() -> const float* { return (const float*)d_in[pi++]; };
  auto getLin = [&]() { Lin l; l.w = P(); l.b = P(); return l; };
  auto getMlp = [&]() { Mlp3 m; m.l0 = getLin(); m.l1 = getLin(); m.l2 = getLin(); return m; };
  auto getLn  = [&]() { LnP l; l.g = P(); l.b = P(); return l; };
  auto getIpa = [&]() {
    IpaP p; p.q = getLin(); p.k = getLin(); p.v = getLin();
    p.qp = getLin(); p.kp = getLin(); p.vp = getLin();
    p.bz = getLin(); p.gamma = P(); p.out = getLin(); return p;
  };

  const float* time_w = P();
  Mlp3 en = getMlp(); LnP enln = getLn();
  LayerP layers[2];
  for (int L = 0; L < 2; ++L) {
    LayerP& lp = layers[L];
    lp.ee = getMlp(); lp.eeln = getLn();
    lp.seu = getMlp(); lp.seln = getLn();
    lp.sp = getIpa(); lp.ln1 = getLn();
    lp.etn = getLin(); lp.et0 = getLin(); lp.et1 = getLin(); lp.etln = getLn();
    lp.sq = getIpa(); lp.ln2 = getLn();
    lp.bb = getLin(); lp.nt = getMlp(); lp.ntln = getLn();
  }
  Mlp3 ft = getMlp(); LnP ftln = getLn();
  Lin tors = getLin(); Lin slog = getLin();

  // ---- workspace bump allocator ----
  size_t woff = 0;
  auto allocf = [&](size_t n) -> float* {
    float* p = (float*)((char*)d_ws + woff);
    woff += ((n * sizeof(float) + 255) / 256) * 256;
    return p;
  };
  auto allocbf = [&](size_t n) -> __bf16* {
    __bf16* p = (__bf16*)((char*)d_ws + woff);
    woff += ((n * sizeof(__bf16) + 255) / 256) * 256;
    return p;
  };
  auto alloci = [&](size_t n) -> int* {
    int* p = (int*)((char*)d_ws + woff);
    woff += ((n * sizeof(int) + 255) / 256) * 256;
    return p;
  };

  const int NV = NNODE;
  const long ESP = (long)NV * EPN_SP;
  const long ESQ = (long)NV * EPN_SEQ;

  float* quats = allocf(NV*4);
  float* trans = allocf(NV*3);
  float* cacc  = allocf(4);
  float* nodein = allocf((size_t)NV*NODE_IN);
  float* s   = allocf((size_t)NV*CS);
  float* h1  = allocf((size_t)NV*CS);
  float* h2  = allocf((size_t)NV*CS);
  float* upd = allocf((size_t)NV*CS);
  float* smask = allocf((size_t)NV*CS);
  int*   seqdst = alloci(ESQ);
  float* sef   = allocf((size_t)ESQ*CZ);
  float* senew = allocf((size_t)ESQ*EF_RAW_D);
  float* seqin = allocf((size_t)ESQ*EIN);
  float* seqh1 = allocf((size_t)ESQ*CZ);
  float* seqh2 = allocf((size_t)ESQ*CZ);
  float* seqmo = allocf((size_t)ESQ*CZ);
  int*   spdst = alloci(ESP);
  float* efraw = allocf((size_t)ESP*EF_RAW_D);
  float* efin  = allocf((size_t)ESP*EIN);
  float* efh1  = allocf((size_t)ESP*CZ);
  float* efh2  = allocf((size_t)ESP*CZ);
  float* ef    = allocf((size_t)ESP*CZ);
  float* qb  = allocf((size_t)NV*NH*CH);
  float* kb  = allocf((size_t)NV*NH*CH);
  float* vb  = allocf((size_t)NV*NH*CH);
  float* qpl = allocf((size_t)NV*NH*PQn*3);
  float* kpl = allocf((size_t)NV*NH*PQn*3);
  float* vpl = allocf((size_t)NV*NH*PVn*3);
  float* qpg = allocf((size_t)NV*NH*PQn*3);
  float* kpg = allocf((size_t)NV*NH*PQn*3);
  float* vpg = allocf((size_t)NV*NH*PVn*3);
  float* bze  = allocf((size_t)ESP*NH);
  float* attn = allocf((size_t)ESP*NH);
  float* feat = allocf((size_t)NV*OUT_FEAT);
  float* hb   = allocf((size_t)NV*(CS/2));
  float* etin = allocf((size_t)ESQ*(CZ + CS));
  float* eth  = allocf((size_t)ESQ*CZ);
  float* bb6  = allocf((size_t)NV*6);
  float* torsin = allocf((size_t)NV*(CS + N_AA));
  // bf16 staging buffers for WMMA GEMM (max A: ESP x EIN, max W: OUT_FEAT x CS)
  __bf16* Abf = allocbf((size_t)ESP * EIN);
  __bf16* Wbf = allocbf((size_t)OUT_FEAT * CS);

  auto BLK = [](long n) { return dim3((unsigned)((n + 255) / 256)); };
  auto gemm = [&](const float* A, const float* W, const float* bias, float* C,
                  long M, int Nn, int K, int act) {
    int Kp = (K + 7) & ~7;   // pad rows to 16B so every 8-elem chunk is aligned
    k_cvt_bf16_pad<<<BLK(M * (long)Kp), 256, 0, stream>>>(A, Abf, M, K, Kp);
    k_cvt_bf16<<<BLK((long)K * Nn), 256, 0, stream>>>(W, Wbf, (long)K * Nn);
    dim3 g((unsigned)((Nn + TILE_N - 1) / TILE_N),
           (unsigned)((M + TILE_M - 1) / TILE_M));
    k_gemm_wmma<<<g, dim3(256), 0, stream>>>(Abf, Wbf, bias, C, (int)M, Nn, K, Kp, act);
  };

  // ---- init rigids + centering ----
  k_quat_norm<<<BLK(NV), 256, 0, stream>>>(rigids, quats, trans);
  k_zero<<<BLK(4), 256, 0, stream>>>(cacc, 4);
  k_center_reduce<<<BLK(NV), 256, 0, stream>>>(trans, res_mask, cacc);
  k_center_apply<<<BLK((long)NV*3), 256, 0, stream>>>(trans, cacc);

  // ---- node embedding ----
  k_node_input<<<BLK((long)NV*NODE_IN), 256, 0, stream>>>(seq_probs, noise_mask, time_w, tptr, nodein);
  gemm(nodein, en.l0.w, en.l0.b, h1, NV, CS, NODE_IN, 1);
  gemm(h1, en.l1.w, en.l1.b, h2, NV, CS, CS, 1);
  gemm(h2, en.l2.w, en.l2.b, upd, NV, CS, CS, 0);
  k_layernorm<<<NV, 256, 0, stream>>>(upd, nullptr, nullptr, enln.g, enln.b, res_mask, s, CS);

  k_seq_build<<<BLK(ESQ), 256, 0, stream>>>(seqdst);
  k_zero<<<BLK(ESQ*CZ), 256, 0, stream>>>(sef, ESQ*CZ);

  auto run_ipa = [&](const IpaP& p, const float* zfeat, const int* edst,
                     int EPN, long E, float* updout) {
    gemm(s, p.q.w, p.q.b, qb, NV, NH*CH, CS, 0);
    gemm(s, p.k.w, p.k.b, kb, NV, NH*CH, CS, 0);
    gemm(s, p.v.w, p.v.b, vb, NV, NH*CH, CS, 0);
    gemm(s, p.qp.w, p.qp.b, qpl, NV, NH*PQn*3, CS, 0);
    gemm(s, p.kp.w, p.kp.b, kpl, NV, NH*PQn*3, CS, 0);
    gemm(s, p.vp.w, p.vp.b, vpl, NV, NH*PVn*3, CS, 0);
    k_pts_glob<<<BLK((long)NV*NH*PQn), 256, 0, stream>>>(quats, trans, qpl, qpg, NH*PQn);
    k_pts_glob<<<BLK((long)NV*NH*PQn), 256, 0, stream>>>(quats, trans, kpl, kpg, NH*PQn);
    k_pts_glob<<<BLK((long)NV*NH*PVn), 256, 0, stream>>>(quats, trans, vpl, vpg, NH*PVn);
    gemm(zfeat, p.bz.w, p.bz.b, bze, E, NH, CZ, 0);
    k_ipa_logits<<<BLK(E*NH), 256, 0, stream>>>(qb, kb, qpg, kpg, bze, p.gamma,
                                                res_mask, edst, EPN, E, attn);
    k_ipa_softmax<<<BLK((long)NV*NH), 256, 0, stream>>>(attn, EPN);
    k_ipa_accum<<<NV, 256, 0, stream>>>(attn, vb, vpg, zfeat, edst, quats, trans, EPN, feat);
    gemm(feat, p.out.w, p.out.b, updout, NV, CS, OUT_FEAT, 0);
  };

  for (int L = 0; L < 2; ++L) {
    const LayerP& lp = layers[L];
    // dynamic spatial graph + edge features
    k_knn<<<BLK(NV), 256, 0, stream>>>(trans, spdst, L);
    k_edge_feats<<<BLK(ESP*EF_RAW_D), 256, 0, stream>>>(quats, trans, spdst, EPN_SP, ESP, efraw);
    k_edge_feats<<<BLK(ESQ*EF_RAW_D), 256, 0, stream>>>(quats, trans, seqdst, EPN_SEQ, ESQ, senew);
    // spatial edge embed MLP + LN
    k_gather_concat<<<BLK(ESP*EIN), 256, 0, stream>>>(efraw, EF_RAW_D, s, CS, spdst, EPN_SP, ESP, efin);
    gemm(efin, lp.ee.l0.w, lp.ee.l0.b, efh1, ESP, CZ, EIN, 1);
    gemm(efh1, lp.ee.l1.w, lp.ee.l1.b, efh2, ESP, CZ, CZ, 1);
    gemm(efh2, lp.ee.l2.w, lp.ee.l2.b, efh1, ESP, CZ, CZ, 0);
    k_layernorm<<<(unsigned)ESP, 256, 0, stream>>>(efh1, nullptr, nullptr, lp.eeln.g, lp.eeln.b, nullptr, ef, CZ);
    // seq edge update MLP + residual LN
    k_gather_concat<<<BLK(ESQ*EIN), 256, 0, stream>>>(senew, EF_RAW_D, s, CS, seqdst, EPN_SEQ, ESQ, seqin);
    gemm(seqin, lp.seu.l0.w, lp.seu.l0.b, seqh1, ESQ, CZ, EIN, 1);
    gemm(seqh1, lp.seu.l1.w, lp.seu.l1.b, seqh2, ESQ, CZ, CZ, 1);
    gemm(seqh2, lp.seu.l2.w, lp.seu.l2.b, seqmo, ESQ, CZ, CZ, 0);
    k_layernorm<<<(unsigned)ESQ, 256, 0, stream>>>(seqmo, sef, nullptr, lp.seln.g, lp.seln.b, nullptr, sef, CZ);
    // IPA over spatial edges
    run_ipa(lp.sp, ef, spdst, EPN_SP, ESP, upd);
    k_layernorm<<<NV, 256, 0, stream>>>(upd, s, res_mask, lp.ln1.g, lp.ln1.b, nullptr, s, CS);
    // IPA over sequence edges
    run_ipa(lp.sq, sef, seqdst, EPN_SEQ, ESQ, upd);
    k_layernorm<<<NV, 256, 0, stream>>>(upd, s, res_mask, lp.ln2.g, lp.ln2.b, nullptr, s, CS);
    // node transition
    gemm(s, lp.nt.l0.w, lp.nt.l0.b, h1, NV, CS, CS, 1);
    gemm(h1, lp.nt.l1.w, lp.nt.l1.b, h2, NV, CS, CS, 1);
    gemm(h2, lp.nt.l2.w, lp.nt.l2.b, upd, NV, CS, CS, 0);
    k_layernorm<<<NV, 256, 0, stream>>>(upd, s, nullptr, lp.ntln.g, lp.ntln.b, res_mask, s, CS);
    // backbone update -> rigids
    k_rowmul<<<BLK((long)NV*CS), 256, 0, stream>>>(s, noise_mask, smask, CS);
    gemm(smask, lp.bb.w, lp.bb.b, bb6, NV, 6, CS, 0);
    k_update_rigids<<<BLK(NV), 256, 0, stream>>>(bb6, noise_mask, quats, trans);
    // edge transition on sequence edges
    gemm(s, lp.etn.w, lp.etn.b, hb, NV, CS/2, CS, 0);
    k_gather_concat<<<BLK(ESQ*(CZ+CS)), 256, 0, stream>>>(sef, CZ, hb, CS/2, seqdst, EPN_SEQ, ESQ, etin);
    gemm(etin, lp.et0.w, lp.et0.b, eth, ESQ, CZ, CZ + CS, 1);
    gemm(eth, lp.et1.w, lp.et1.b, seqmo, ESQ, CZ, CZ, 0);
    k_layernorm<<<(unsigned)ESQ, 256, 0, stream>>>(seqmo, nullptr, nullptr, lp.etln.g, lp.etln.b, nullptr, sef, CZ);
  }

  // ---- final transition + heads ----
  float* out_s = (float*)d_out;
  float* out_r = out_s + (size_t)NV*CS;
  float* out_t = out_r + (size_t)NV*7;
  float* out_l = out_t + (size_t)NV*162;

  gemm(s, ft.l0.w, ft.l0.b, h1, NV, CS, CS, 1);
  gemm(h1, ft.l1.w, ft.l1.b, h2, NV, CS, CS, 1);
  gemm(h2, ft.l2.w, ft.l2.b, upd, NV, CS, CS, 0);
  k_layernorm<<<NV, 256, 0, stream>>>(upd, s, nullptr, ftln.g, ftln.b, nullptr, out_s, CS);

  k_write_rigids<<<BLK(NV), 256, 0, stream>>>(quats, trans, out_r);
  k_concat2<<<BLK((long)NV*(CS+N_AA)), 256, 0, stream>>>(out_s, CS, seq_probs, N_AA, torsin);
  gemm(torsin, tors.w, tors.b, out_t, NV, 162, CS + N_AA, 0);
  gemm(out_s, slog.w, slog.b, out_l, NV, N_AA, CS, 0);
}
